// InfoNCE_42511586296568
// MI455X (gfx1250) — compile-verified, matrix operand order checked by reading, not run on hardware
//
#include <hip/hip_runtime.h>
#include <math.h>

typedef unsigned short u16;
typedef unsigned long long u64;
typedef __attribute__((ext_vector_type(16))) __bf16 v16bf;
typedef __attribute__((ext_vector_type(8)))  __bf16 v8bf;
typedef __attribute__((ext_vector_type(8)))  float  v8f;

#define BDIM 4096
#define MDIM 16384
#define CDIM 256
#define NTILES ((BDIM + MDIM) / 16)   // 1280 column tiles of 16
#define BANK_TILE0 (BDIM / 16)        // first bank column tile
#define CCHUNKS 4                     // column-tile chunks (parallelism)
#define CT_PER (NTILES / CCHUNKS)     // 320 tiles per chunk

__device__ __forceinline__ u16 f32_to_bf16(float x) {
  unsigned int u = __float_as_uint(x);
  u += 0x7fffu + ((u >> 16) & 1u);    // round-to-nearest-even
  return (u16)(u >> 16);
}

// 16xK=32 WMMA bf16 fragment per the CDNA5 16-bit A/B layout:
// lane elems 0..7 = K[k0 + half*8 ..], elems 8..15 = K[k0+16+half*8 ..]
__device__ __forceinline__ v16bf make_frag(const u16* p0, const u16* p1) {
  const v8bf x = *(const v8bf*)p0;
  const v8bf y = *(const v8bf*)p1;
  v16bf r;
#pragma unroll
  for (int i = 0; i < 8; ++i) { r[i] = x[i]; r[i + 8] = y[i]; }
  return r;
}

// acc layout (floats): [0]=ce_sum  [1],[2]=icel sq-sums  (int)[3],[4]=icel counts
__global__ void init_kernel(float* acc, float* rowsum, u64* rowkey) {
  const int i = blockIdx.x * blockDim.x + threadIdx.x;
  if (i < 8) acc[i] = 0.0f;
  if (i < 2 * BDIM) { rowsum[i] = 0.0f; rowkey[i] = 0ull; }
}

// One wave per row: L2-normalize f1/f2, emit f32 + bf16 copies, and the
// diagonal logit scale*<f1n_i, f2n_i> (identical for both CE directions).
__global__ void normalize_kernel(const float* __restrict__ f1,
                                 const float* __restrict__ f2,
                                 const float* __restrict__ scale_ptr,
                                 float* __restrict__ f1n, float* __restrict__ f2n,
                                 u16* __restrict__ f1b, u16* __restrict__ f2b,
                                 float* __restrict__ diag) {
  const int lane = threadIdx.x & 31;
  const int wave = threadIdx.x >> 5;
  const int row  = blockIdx.x * (blockDim.x >> 5) + wave;
  if (row >= BDIM) return;
  const int base = row * CDIM + lane * 8;
  float a[8], b[8];
#pragma unroll
  for (int i = 0; i < 8; ++i) { a[i] = f1[base + i]; b[i] = f2[base + i]; }
  float sa = 0.f, sb = 0.f;
#pragma unroll
  for (int i = 0; i < 8; ++i) { sa += a[i] * a[i]; sb += b[i] * b[i]; }
#pragma unroll
  for (int off = 1; off < 32; off <<= 1) {
    sa += __shfl_xor(sa, off, 32);
    sb += __shfl_xor(sb, off, 32);
  }
  const float ia = 1.0f / fmaxf(sqrtf(sa), 1e-12f);
  const float ib = 1.0f / fmaxf(sqrtf(sb), 1e-12f);
  float dot = 0.f;
#pragma unroll
  for (int i = 0; i < 8; ++i) {
    const float an = a[i] * ia, bn = b[i] * ib;
    f1n[base + i] = an;              f2n[base + i] = bn;
    f1b[base + i] = f32_to_bf16(an); f2b[base + i] = f32_to_bf16(bn);
    dot += an * bn;
  }
#pragma unroll
  for (int off = 1; off < 32; off <<= 1) dot += __shfl_xor(dot, off, 32);
  if (lane == 0) diag[row] = scale_ptr[0] * dot;
}

__global__ void bank_convert_kernel(const float* __restrict__ mb1,
                                    const float* __restrict__ mb2,
                                    u16* __restrict__ b1, u16* __restrict__ b2) {
  const int i = blockIdx.x * blockDim.x + threadIdx.x;
  if (i < MDIM * CDIM) { b1[i] = f32_to_bf16(mb1[i]); b2[i] = f32_to_bf16(mb2[i]); }
}

// Fused logits kernel. Block = 4 waves, each wave owns one 16-row tile of A;
// block marches over its chunk of 320 column tiles of [f_other ; bank],
// async-staged (GLOBAL_LOAD_ASYNC_TO_LDS_B128) into double-buffered LDS.
// Tile body: 16 ds_load_b128 -> sched_barrier -> 8 back-to-back WMMAs on two
// independent accumulator chains (even/odd K) -> epilogue VALU.
__launch_bounds__(128)
__global__ void logits_kernel(const u16* __restrict__ f1b, const u16* __restrict__ f2b,
                              const u16* __restrict__ mb1b, const u16* __restrict__ mb2b,
                              const float* __restrict__ scale_ptr,
                              float* __restrict__ rowsum,
                              u64*   __restrict__ rowkey) {
  const int dir = blockIdx.z;
  const u16* __restrict__ Af    = dir ? f2b : f1b;
  const u16* __restrict__ Bfrnt = dir ? f1b : f2b;
  const u16* __restrict__ Bbank = dir ? mb1b : mb2b;
  const float scale = scale_ptr[0];

  const int lane = threadIdx.x & 31;
  const int wave = threadIdx.x >> 5;
  const int half = lane >> 4;
  const int lrow = lane & 15;
  const int rt   = blockIdx.x * 4 + wave;          // row tile 0..255
  const int c0   = blockIdx.y * CT_PER;
  const int cend = c0 + CT_PER;

  __shared__ __align__(16) u16 Bt[2][16 * CDIM];   // 2 x 8KB double buffer

  // Preload this wave's A fragments: 16 rows x K=256 -> 8 fragments of K=32
  const u16* arow = Af + (size_t)(rt * 16 + lrow) * CDIM;
  v16bf afrag[8];
#pragma unroll
  for (int k = 0; k < 8; ++k)
    afrag[k] = make_frag(arow + k * 32 + half * 8, arow + k * 32 + 16 + half * 8);

  float sumexp[8]; float bmax[8]; int barg[8];
#pragma unroll
  for (int r = 0; r < 8; ++r) { sumexp[r] = 0.f; bmax[r] = -2.0f; barg[r] = 0; }

  // Async staging: 128 threads x 64B = 8KB tile, direct global -> LDS.
  const int crow = threadIdx.x >> 3;               // 0..15 : tile row
  const int ccol = (threadIdx.x & 7) * 32;         // 0..224: 32 elems / thread
  auto stage = [&](int ct, int buf) {
    const int grow = ct * 16 + crow;
    const u16* src = (grow < BDIM) ? (Bfrnt + (size_t)grow * CDIM + ccol)
                                   : (Bbank + (size_t)(grow - BDIM) * CDIM + ccol);
    // LDS byte offset = low 32 bits of the generic pointer (aperture mapping)
    const unsigned ldsoff =
        (unsigned)(unsigned long long)(void*)&Bt[buf][crow * CDIM + ccol];
#pragma unroll
    for (int j = 0; j < 4; ++j)
      asm volatile("global_load_async_to_lds_b128 %0, %1, off offset:%2"
                   :: "v"(ldsoff), "v"(src), "i"(j * 16) : "memory");
  };

  stage(c0, 0);
  asm volatile("s_wait_asynccnt 0x0" ::: "memory");
  __syncthreads();

  for (int ct = c0; ct < cend; ++ct) {
    const int buf = (ct - c0) & 1;
    if (ct + 1 < cend) stage(ct + 1, buf ^ 1);

    const u16* brow = &Bt[buf][lrow * CDIM];
    v16bf bfrag[8];
#pragma unroll
    for (int k = 0; k < 8; ++k)
      bfrag[k] = make_frag(brow + k * 32 + half * 8,
                           brow + k * 32 + 16 + half * 8);

    // Keep all 16 ds_loads before, all WMMAs after: one dscnt wait, then a
    // clean burst of 8 WMMAs on two independent accumulator chains.
    __builtin_amdgcn_sched_barrier(0);

    v8f ce = {}, co = {};
#pragma unroll
    for (int k = 0; k < 8; k += 2) {
      ce = __builtin_amdgcn_wmma_f32_16x16x32_bf16(false, afrag[k],     false, bfrag[k],
                                                   (short)0, ce, false, false);
      co = __builtin_amdgcn_wmma_f32_16x16x32_bf16(false, afrag[k + 1], false, bfrag[k + 1],
                                                   (short)0, co, false, false);
    }

    const bool isbank = (ct >= BANK_TILE0);
    const int  idx    = (ct - BANK_TILE0) * 16 + lrow;
#pragma unroll
    for (int r = 0; r < 8; ++r) {
      const float l = ce[r] + co[r];               // raw dot (row r+half*8, col lrow)
      sumexp[r] += __expf(scale * l);
      if (isbank && l > bmax[r]) { bmax[r] = l; barg[r] = idx; }
    }
    asm volatile("s_wait_asynccnt 0x0" ::: "memory");
    __syncthreads();
  }

  // Reduce across the 16 lanes of each half (columns), per row slot.
#pragma unroll
  for (int r = 0; r < 8; ++r) {
    float s = sumexp[r], m = bmax[r]; int a = barg[r];
#pragma unroll
    for (int off = 1; off < 16; off <<= 1) {
      s += __shfl_xor(s, off, 32);
      const float om = __shfl_xor(m, off, 32);
      const int   oa = __shfl_xor(a, off, 32);
      if (om > m || (om == m && oa < a)) { m = om; a = oa; }
    }
    if (lrow == 0) {
      const int row = rt * 16 + half * 8 + r;
      atomicAdd(&rowsum[dir * BDIM + row], s);
      // monotone key: bits(m+2) (m+2 in [1,3] > 0), ~idx so ties pick lowest idx
      const u64 pk = ((u64)__float_as_uint(m + 2.0f) << 32) |
                     (u64)(0xFFFFFFFFu - (unsigned)a);
      atomicMax(&rowkey[dir * BDIM + row], pk);
    }
  }
}

// Per-row CE term: log(sum exp) - diag; block-reduced, one atomic per block.
__global__ void ce_reduce_kernel(const float* __restrict__ rowsum,
                                 const float* __restrict__ diag,
                                 float* __restrict__ acc) {
  const int i = blockIdx.x * blockDim.x + threadIdx.x;   // 0 .. 2*BDIM-1
  float v = 0.f;
  if (i < 2 * BDIM) v = __logf(rowsum[i]) - diag[i & (BDIM - 1)];
#pragma unroll
  for (int off = 1; off < 32; off <<= 1) v += __shfl_xor(v, off, 32);
  __shared__ float wsum[8];
  const int lane = threadIdx.x & 31, wv = threadIdx.x >> 5;
  if (lane == 0) wsum[wv] = v;
  __syncthreads();
  if (threadIdx.x < 8) {
    float t = wsum[threadIdx.x];
#pragma unroll
    for (int off = 1; off < 8; off <<= 1) t += __shfl_xor(t, off, 32);
    if (threadIdx.x == 0) atomicAdd(&acc[0], t);
  }
}

// One wave per (dir,row): masked MSE vs gathered nearest-neighbor bank row.
__global__ void icel_kernel(const float* __restrict__ f1n, const float* __restrict__ f2n,
                            const float* __restrict__ mb1, const float* __restrict__ mb2,
                            const u64* __restrict__ rowkey,
                            float* __restrict__ acc) {
  const int dir = blockIdx.y;
  const int row = blockIdx.x;
  const u64 k = rowkey[dir * BDIM + row];
  const float m = __uint_as_float((unsigned)(k >> 32)) - 2.0f;
  if (!(m > 0.2f)) return;
  const int nn = (int)(0xFFFFFFFFu - (unsigned)k);
  const float* f    = (dir ? f2n : f1n) + (size_t)row * CDIM;
  const float* bank = (dir ? mb1 : mb2) + (size_t)nn * CDIM;
  const int lane = threadIdx.x;
  float sq = 0.f;
#pragma unroll
  for (int i = 0; i < 8; ++i) {
    const float d = f[lane * 8 + i] - bank[lane * 8 + i];
    sq += d * d;
  }
#pragma unroll
  for (int off = 1; off < 32; off <<= 1) sq += __shfl_xor(sq, off, 32);
  if (lane == 0) {
    atomicAdd(&acc[1 + dir], sq);
    atomicAdd((int*)acc + 3 + dir, 1);
  }
}

__global__ void finalize_kernel(const float* __restrict__ acc, float* __restrict__ out) {
  if (threadIdx.x == 0) {
    const float nce = acc[0] / (2.0f * (float)BDIM);   // 0.5*(mean CE1 + mean CE2)
    const int c1 = ((const int*)acc)[3];
    const int c2 = ((const int*)acc)[4];
    const float icel1 = (c1 > 0) ? acc[1] / ((float)c1 * (float)CDIM) : 0.f;
    const float icel2 = (c2 > 0) ? acc[2] / ((float)c2 * (float)CDIM) : 0.f;
    out[0] = nce + 0.5f * (0.5f * (icel1 + icel2));    // LAMBDA_ICEL = 0.5
  }
}

extern "C" void kernel_launch(void* const* d_in, const int* in_sizes, int n_in,
                              void* d_out, int out_size, void* d_ws, size_t ws_size,
                              hipStream_t stream) {
  (void)in_sizes; (void)n_in; (void)out_size; (void)ws_size;
  const float* f1  = (const float*)d_in[0];
  const float* f2  = (const float*)d_in[1];
  const float* sc  = (const float*)d_in[2];
  const float* mb1 = (const float*)d_in[3];
  const float* mb2 = (const float*)d_in[4];

  char* ws = (char*)d_ws;
  size_t off = 0;
  auto alloc = [&](size_t n) -> void* {
    void* p = ws + off;
    off += (n + 255) & ~(size_t)255;
    return p;
  };
  float* f1n    = (float*)alloc((size_t)BDIM * CDIM * 4);
  float* f2n    = (float*)alloc((size_t)BDIM * CDIM * 4);
  u16*   f1b    = (u16*)  alloc((size_t)BDIM * CDIM * 2);
  u16*   f2b    = (u16*)  alloc((size_t)BDIM * CDIM * 2);
  u16*   b1b    = (u16*)  alloc((size_t)MDIM * CDIM * 2);
  u16*   b2b    = (u16*)  alloc((size_t)MDIM * CDIM * 2);
  float* diag   = (float*)alloc((size_t)BDIM * 4);
  float* rowsum = (float*)alloc((size_t)2 * BDIM * 4);
  u64*   rowkey = (u64*)  alloc((size_t)2 * BDIM * 8);
  float* acc    = (float*)alloc(256);

  init_kernel<<<(2 * BDIM + 255) / 256, 256, 0, stream>>>(acc, rowsum, rowkey);
  normalize_kernel<<<BDIM / 8, 256, 0, stream>>>(f1, f2, sc, f1n, f2n, f1b, f2b, diag);
  bank_convert_kernel<<<(MDIM * CDIM + 255) / 256, 256, 0, stream>>>(mb1, mb2, b1b, b2b);
  logits_kernel<<<dim3(BDIM / 64, CCHUNKS, 2), 128, 0, stream>>>(f1b, f2b, b1b, b2b,
                                                                 sc, rowsum, rowkey);
  ce_reduce_kernel<<<(2 * BDIM + 255) / 256, 256, 0, stream>>>(rowsum, diag, acc);
  icel_kernel<<<dim3(BDIM, 2), 32, 0, stream>>>(f1n, f2n, mb1, mb2, rowkey, acc);
  finalize_kernel<<<1, 32, 0, stream>>>(acc, (float*)d_out);
}
